// ShiftedPatchTokenization_12369505812905
// MI455X (gfx1250) — matrix operand order repaired and checked by reading, not compiled, not run on hardware
//
#include <hip/hip_runtime.h>
#include <hip/hip_bf16.h>

typedef __bf16 bf16;
typedef __attribute__((ext_vector_type(16))) bf16  v16bf;
typedef __attribute__((ext_vector_type(8)))  bf16  v8bf;
typedef __attribute__((ext_vector_type(8)))  float v8f;

#define IMG    256
#define PATCH  16
#define BATCH  32
#define M_TOT  8192   // BATCH * 256 tokens
#define K_TOT  3840   // 16*16*3*5
#define N_TOT  768

// ---------------------------------------------------------------------------
// Kernel 1: shifted-patch gather + LayerNorm (fp32 stats) + bf16 A to ws
// ---------------------------------------------------------------------------
__global__ __launch_bounds__(256) void spt_patch_ln(
    const float* __restrict__ images, const float* __restrict__ gamma,
    const float* __restrict__ beta,   float* __restrict__ patches,
    bf16* __restrict__ Abf)
{
  const int tid   = threadIdx.x;
  const int token = blockIdx.x;        // 0..8191
  const int b  = token >> 8;
  const int t  = token & 255;
  const int ph = t >> 4, pw = t & 15;

  float v[15];
  float sum = 0.f, sq = 0.f;
#pragma unroll
  for (int i = 0; i < 15; ++i) {
    int d   = i * 256 + tid;           // 0..3839 (coalesced in d)
    int pix = d / 15;
    int c5  = d - pix * 15;
    int py = pix >> 4, px = pix & 15;
    int y = ph * PATCH + py, x = pw * PATCH + px;
    int vi = c5 / 3;
    int ch = c5 - vi * 3;
    int sy = y, sx = x; bool ok = true;
    if      (vi == 1) { sy = y + 8; sx = x + 8; ok = (y < 248) && (x < 248); }
    else if (vi == 2) { sy = y - 8; sx = x + 8; ok = (y >= 8)  && (x < 248); }
    else if (vi == 3) { sy = y + 8; sx = x - 8; ok = (y < 248) && (x >= 8);  }
    else if (vi == 4) { sy = y - 8; sx = x - 8; ok = (y >= 8)  && (x >= 8);  }
    float val = ok ? images[(((size_t)b * IMG + sy) * IMG + sx) * 3 + ch] : 0.f;
    v[i] = val; sum += val; sq += val * val;
  }

  __shared__ float rs[256], rq[256];
  rs[tid] = sum; rq[tid] = sq;
  __syncthreads();
  for (int s = 128; s > 0; s >>= 1) {
    if (tid < s) { rs[tid] += rs[tid + s]; rq[tid] += rq[tid + s]; }
    __syncthreads();
  }
  const float mean = rs[0] * (1.f / 3840.f);
  const float var  = rq[0] * (1.f / 3840.f) - mean * mean;
  const float rstd = rsqrtf(var + 1e-5f);

  const size_t base = (size_t)token * K_TOT;
#pragma unroll
  for (int i = 0; i < 15; ++i) {
    int d = i * 256 + tid;
    patches[base + d] = v[i];
    float nrm = (v[i] - mean) * rstd * gamma[d] + beta[d];
    Abf[base + d] = (bf16)nrm;
  }
}

// ---------------------------------------------------------------------------
// Kernel 2: W (3840x768 f32, row-major) -> Wt (768x3840 bf16) via LDS tiles
// ---------------------------------------------------------------------------
__global__ __launch_bounds__(256) void spt_transposeW(
    const float* __restrict__ W, bf16* __restrict__ Wt)
{
  __shared__ float tile[32][33];
  const int k0 = blockIdx.x * 32, n0 = blockIdx.y * 32;
  const int tx = threadIdx.x, ty = threadIdx.y;   // 32 x 8
#pragma unroll
  for (int j = 0; j < 32; j += 8)
    tile[ty + j][tx] = W[(size_t)(k0 + ty + j) * N_TOT + n0 + tx];
  __syncthreads();
#pragma unroll
  for (int j = 0; j < 32; j += 8)
    Wt[(size_t)(n0 + ty + j) * K_TOT + k0 + tx] = (bf16)tile[tx][ty + j];
}

// ---------------------------------------------------------------------------
// Kernel 3: GEMM  tokens = A(8192x3840 bf16) * Wt^T + bias, f32 accumulate.
// 128x128 block tile, 8 waves x (32x64), BK=32, double-buffered LDS fed by
// GLOBAL_LOAD_ASYNC_TO_LDS_B128 (ASYNCcnt).  Double-step main loop so buffer
// indices are compile-time constants -> branch-free, cndmask-free hot loop.
// ---------------------------------------------------------------------------
#define BM 128
#define BN 128
#define BK 32
#define LDT 40   // padded LDS row: 40 bf16 = 80 bytes (conflict-free b128)

__device__ __forceinline__ void async_b128(unsigned lds_off, const bf16* g) {
  asm volatile("global_load_async_to_lds_b128 %0, %1, off"
               :: "v"(lds_off), "v"((unsigned long long)(uintptr_t)g)
               : "memory");
}

__global__ __launch_bounds__(256) void spt_gemm(
    const bf16* __restrict__ A, const bf16* __restrict__ Wt,
    const float* __restrict__ bias, float* __restrict__ out)
{
  __shared__ __align__(16) bf16 sA[2][BM * LDT];
  __shared__ __align__(16) bf16 sB[2][BN * LDT];

  const int tid  = threadIdx.x;
  const int lane = tid & 31;
  const int wave = tid >> 5;      // 0..7
  const int wm   = wave & 3;      // 4 waves along M (32 rows each)
  const int wn   = wave >> 2;     // 2 waves along N (64 cols each)

  const int bm0 = blockIdx.x * BM;
  const int bn0 = blockIdx.y * BN;

  const int lrow = tid >> 1;      // 0..127 : staged row
  const int lhal = tid & 1;       // 0/1    : 32-byte half of the 64B K-row
  const int lmod = lane & 15;
  const int lhi  = lane >> 4;     // 0/1

  v8f acc[2][4];
#pragma unroll
  for (int mi = 0; mi < 2; ++mi)
#pragma unroll
    for (int ni = 0; ni < 4; ++ni) acc[mi][ni] = v8f{};

  // per-thread global sources (32 B of A-tile + 32 B of B-tile per K-step)
  const bf16* gA = A  + (size_t)(bm0 + lrow) * K_TOT + lhal * 16;
  const bf16* gB = Wt + (size_t)(bn0 + lrow) * K_TOT + lhal * 16;

  // per-thread LDS destinations (byte offsets; ISA: LDS addr = addr[31:0])
  unsigned dA[2], dB[2];
  dA[0] = (unsigned)(uintptr_t)&sA[0][lrow * LDT + lhal * 16];
  dA[1] = (unsigned)(uintptr_t)&sA[1][lrow * LDT + lhal * 16];
  dB[0] = (unsigned)(uintptr_t)&sB[0][lrow * LDT + lhal * 16];
  dB[1] = (unsigned)(uintptr_t)&sB[1][lrow * LDT + lhal * 16];

  // issue one K-step's tiles (4 async b128 per thread) into buffer `buf`
  auto issue = [&](int buf, int koff) {
    const bf16* ga = gA + koff;
    const bf16* gb = gB + koff;
    async_b128(dA[buf],      ga);
    async_b128(dA[buf] + 16, ga + 8);
    async_b128(dB[buf],      gb);
    async_b128(dB[buf] + 16, gb + 8);
  };

  // consume one K-step from buffer `buf`: 12 ds_load_b128 + 8 WMMA
  auto compute = [&](int buf) {
    const bf16* cA = sA[buf];
    const bf16* cB = sB[buf];
    // A 16x32: lanes 0-15 K{0..7,16..23}, lanes 16-31 K{8..15,24..31}
    v16bf af[2];
#pragma unroll
    for (int mi = 0; mi < 2; ++mi) {
      const bf16* p = &cA[(wm * 32 + mi * 16 + lmod) * LDT + lhi * 8];
      v8bf lo = *(const v8bf*)(p);
      v8bf hi = *(const v8bf*)(p + 16);
      af[mi] = __builtin_shufflevector(lo, hi, 0,1,2,3,4,5,6,7,8,9,10,11,12,13,14,15);
    }
    // B 32x16: lanes 0-15 K{0..7,16..23}, lanes 16-31 K{8..15,24..31} (row n = lane%16)
    v16bf bfv[4];
#pragma unroll
    for (int ni = 0; ni < 4; ++ni) {
      const bf16* p = &cB[(wn * 64 + ni * 16 + lmod) * LDT + lhi * 8];
      v8bf lo = *(const v8bf*)(p);
      v8bf hi = *(const v8bf*)(p + 16);
      bfv[ni] = __builtin_shufflevector(lo, hi, 0,1,2,3,4,5,6,7,8,9,10,11,12,13,14,15);
    }
#pragma unroll
    for (int mi = 0; mi < 2; ++mi)
#pragma unroll
      for (int ni = 0; ni < 4; ++ni)
        acc[mi][ni] = __builtin_amdgcn_wmma_f32_16x16x32_bf16(
            false, af[mi], false, bfv[ni], (short)0, acc[mi][ni], false, false);
  };

  issue(0, 0);   // prologue: fill buffer 0 with K-step 0

  // main loop: two K-steps per iteration, buffers at fixed indices.
  // covers K-steps 0 .. (NSTEP-3); tail handles the last two steps.
  for (int kt = 0; kt < K_TOT - 2 * BK; kt += 2 * BK) {
    issue(1, kt + BK);                                 // next tile -> buf1
    asm volatile("s_wait_asynccnt 0x4" ::: "memory");  // buf0 complete (mine)
    __syncthreads();                                   // complete for all
    compute(0);
    __syncthreads();                                   // buf0 free to refill

    issue(0, kt + 2 * BK);                             // next tile -> buf0
    asm volatile("s_wait_asynccnt 0x4" ::: "memory");
    __syncthreads();
    compute(1);
    __syncthreads();
  }

  // tail: K-steps (NSTEP-2) in buf0 [already issued], (NSTEP-1) into buf1
  issue(1, K_TOT - BK);
  asm volatile("s_wait_asynccnt 0x4" ::: "memory");
  __syncthreads();
  compute(0);
  __syncthreads();

  asm volatile("s_wait_asynccnt 0x0" ::: "memory");
  __syncthreads();
  compute(1);

  // ---- epilogue: C/D layout -> lanes 0-15 N=lmod M=r, lanes 16-31 M=8+r ----
#pragma unroll
  for (int mi = 0; mi < 2; ++mi) {
#pragma unroll
    for (int ni = 0; ni < 4; ++ni) {
      const int gn = bn0 + wn * 64 + ni * 16 + lmod;
      const float bb = bias[gn];
      const int gm = bm0 + wm * 32 + mi * 16 + lhi * 8;
#pragma unroll
      for (int r = 0; r < 8; ++r)
        out[(size_t)(gm + r) * N_TOT + gn] = acc[mi][ni][r] + bb;
    }
  }
}

// ---------------------------------------------------------------------------
extern "C" void kernel_launch(void* const* d_in, const int* in_sizes, int n_in,
                              void* d_out, int out_size, void* d_ws, size_t ws_size,
                              hipStream_t stream) {
  const float* images = (const float*)d_in[0];
  const float* gamma  = (const float*)d_in[1];
  const float* beta   = (const float*)d_in[2];
  const float* W      = (const float*)d_in[3];
  const float* bvec   = (const float*)d_in[4];

  float* tokens  = (float*)d_out;                       // 8192 x 768
  float* patches = tokens + (size_t)M_TOT * N_TOT;      // 8192 x 3840

  bf16* Abf = (bf16*)d_ws;                                                  // 8192x3840
  bf16* Wt  = (bf16*)((char*)d_ws + (size_t)M_TOT * K_TOT * sizeof(bf16));  // 768x3840

  spt_patch_ln<<<M_TOT, 256, 0, stream>>>(images, gamma, beta, patches, Abf);

  dim3 tgrid(K_TOT / 32, N_TOT / 32);
  spt_transposeW<<<tgrid, dim3(32, 8), 0, stream>>>(W, Wt);

  dim3 ggrid(M_TOT / BM, N_TOT / BN);
  spt_gemm<<<ggrid, 256, 0, stream>>>(Abf, Wt, bvec, tokens);
}
